// Qwen3NextMoE_11922829214185
// MI455X (gfx1250) — matrix operand order repaired and checked by reading, not compiled, not run on hardware
//
#include <hip/hip_runtime.h>
#include <hip/hip_bf16.h>

// ---------------------------------------------------------------------------
// MoE (shared + top-2 routed experts), gathered-token formulation.
// Math path: bf16 WMMA (v_wmma_f32_16x16x32_bf16), f32 accumulate, f32 epilogue.
// ---------------------------------------------------------------------------

#define NTOK 1024      // B*T
#define CDIM 1024
#define IDIM 4096
#define NEXP 8
#define TOPK 2
#define LDT  40        // LDS tile row stride in bf16 elems (80B rows, 16B-aligned)

typedef __attribute__((ext_vector_type(16))) __bf16 v16bf;
typedef __attribute__((ext_vector_type(8)))  float  v8f;

union FragB16 { uint4 q[2]; v16bf v; };

// ---------------------------------------------------------------------------
// 128x128 output tile GEMM core, BK=32, 256 threads = 8 waves (wave32).
// Wave (wm 0..3, wn 0..1) owns a 32x64 sub-tile = 2x4 WMMA accumulators.
//   D[row, n0+col] = act( A[row,:] @ W[:, n0+col] + bias[n0+col] )
// A is bf16 row-major [*, Kdim]; W is f32 k-major [Kdim, Ndim] (converted to
// bf16 while staging to LDS). GATHER: A row index comes from gather[].
// ---------------------------------------------------------------------------
template<bool GATHER, bool SILU>
__device__ __forceinline__ void gemm128(
    const __bf16* __restrict__ A, const int* __restrict__ gather,
    int rowbase, int nvalid,
    const float* __restrict__ W, const float* __restrict__ bias,
    void* __restrict__ Out, int Kdim, int Ndim, int n0)
{
  __shared__ __attribute__((aligned(16))) __bf16 As[128 * LDT];
  __shared__ __attribute__((aligned(16))) __bf16 Bs[128 * LDT];

  const int tid  = threadIdx.x;
  const int lane = tid & 31;
  const int wave = tid >> 5;
  const int wm   = wave >> 1;   // 0..3 : 32-row strip
  const int wn   = wave & 1;    // 0..1 : 64-col strip
  const int half = lane >> 4;   // hi/lo 16-lane group
  const int l15  = lane & 15;

  // global->LDS copy roles
  const int arow  = tid >> 1;   // 0..127 (A tile row)
  const int apart = tid & 1;    // which 32B half of the 64B bf16 row chunk
  const int brow  = tid >> 7;   // 0..1   (B k-row phase)
  const int bn    = tid & 127;  // B tile column

  const bool avalid = (arow < nvalid);
  int asrc;                     // source row in A for this thread's copy
  if (GATHER) asrc = avalid ? gather[rowbase + arow] : gather[rowbase];
  else        asrc = avalid ? (rowbase + arow) : rowbase;

  v8f acc[2][4];
  #pragma unroll
  for (int i = 0; i < 2; ++i)
    #pragma unroll
    for (int j = 0; j < 4; ++j)
      #pragma unroll
      for (int k = 0; k < 8; ++k) acc[i][j][k] = 0.0f;

  for (int k0 = 0; k0 < Kdim; k0 += 32) {
    // ---- global loads: A 128x32 bf16 (2x uint4/thread), B 32x128 f32 ----
    const uint4* ap = (const uint4*)(A + (size_t)asrc * Kdim + k0 + apart * 16);
    uint4 aq0 = ap[0];
    uint4 aq1 = ap[1];
    if (!avalid) { aq0 = make_uint4(0,0,0,0); aq1 = make_uint4(0,0,0,0); }

    float bw[16];
    #pragma unroll
    for (int it = 0; it < 16; ++it) {
      int kk = brow + it * 2;   // 0..31
      bw[it] = W[(size_t)(k0 + kk) * Ndim + n0 + bn];
    }
    if (k0 + 32 < Kdim)  // prefetch next W k-slab (-> global_prefetch_b8)
      __builtin_prefetch(&W[(size_t)(k0 + 32 + brow) * Ndim + n0 + bn], 0, 1);

    __syncthreads();  // previous iteration's LDS fragment reads are done

    // ---- stage to LDS (A raw bf16, B converted f32->bf16, column-major) ----
    uint4* ad = (uint4*)&As[arow * LDT + apart * 16];
    ad[0] = aq0; ad[1] = aq1;
    #pragma unroll
    for (int it = 0; it < 16; ++it) {
      int kk = brow + it * 2;
      Bs[bn * LDT + kk] = (__bf16)bw[it];
    }

    __syncthreads();

    // ---- build fragments per ISA 7.12.2 (each = 2x ds_load_b128) ----
    FragB16 afr[2], bfr[4];
    #pragma unroll
    for (int sm = 0; sm < 2; ++sm) {
      int m = wm * 32 + sm * 16 + l15;
      afr[sm].q[0] = *(const uint4*)&As[m * LDT + half * 8];       // K = half*8 .. +7
      afr[sm].q[1] = *(const uint4*)&As[m * LDT + 16 + half * 8];  // K = 16+half*8 .. +7
    }
    #pragma unroll
    for (int sn = 0; sn < 4; ++sn) {
      int n = wn * 64 + sn * 16 + l15;
      bfr[sn].q[0] = *(const uint4*)&Bs[n * LDT + half * 8];       // K = half*8 .. +7
      bfr[sn].q[1] = *(const uint4*)&Bs[n * LDT + 16 + half * 8];  // K = 16+half*8 .. +7
    }

    #pragma unroll
    for (int sm = 0; sm < 2; ++sm)
      #pragma unroll
      for (int sn = 0; sn < 4; ++sn)
        acc[sm][sn] = __builtin_amdgcn_wmma_f32_16x16x32_bf16(
            false, afr[sm].v, false, bfr[sn].v, (short)0, acc[sm][sn],
            false, false);
  }

  // ---- epilogue: +bias, optional SiLU, store (C/D layout: m = i + 8*half) ----
  #pragma unroll
  for (int sn = 0; sn < 4; ++sn) {
    int n = n0 + wn * 64 + sn * 16 + l15;
    float bv = bias[n];
    #pragma unroll
    for (int sm = 0; sm < 2; ++sm) {
      #pragma unroll
      for (int i = 0; i < 8; ++i) {
        int mloc = wm * 32 + sm * 16 + half * 8 + i;
        if (mloc < nvalid) {
          float v = acc[sm][sn][i] + bv;
          size_t orow = (size_t)(rowbase + mloc);
          if constexpr (SILU) {
            v = v / (1.0f + __expf(-v));            // silu(x)=x*sigmoid(x)
            ((__bf16*)Out)[orow * (size_t)Ndim + n] = (__bf16)v;
          } else {
            ((float*)Out)[orow * (size_t)Ndim + n] = v;
          }
        }
      }
    }
  }
}

// ---------------------------------------------------------------------------
// Kernels
// ---------------------------------------------------------------------------

// x -> bf16, zero expert histogram
__global__ void k_init(const float* __restrict__ x, __bf16* __restrict__ xb,
                       int* __restrict__ counts) {
  int i = blockIdx.x * 256 + threadIdx.x;
  xb[i] = (__bf16)x[i];
  if (blockIdx.x == 0 && threadIdx.x < NEXP) counts[threadIdx.x] = 0;
}

// Router MLP + softmax + top-2 + renormalized gates; one block per token.
__global__ void k_router(const float* __restrict__ x,
                         const float* __restrict__ Wr1, const float* __restrict__ br1,
                         const float* __restrict__ Wr2, const float* __restrict__ br2,
                         int* __restrict__ counts, int* __restrict__ topk_e,
                         float* __restrict__ topk_w) {
  __shared__ float xs[CDIM];
  __shared__ float hs[256];
  __shared__ float lg[NEXP];
  const int t = blockIdx.x, tid = threadIdx.x;
  for (int i = tid; i < CDIM; i += 256) xs[i] = x[t * CDIM + i];
  __syncthreads();
  float a = br1[tid];
  for (int k = 0; k < CDIM; ++k) a = fmaf(xs[k], Wr1[k * 256 + tid], a);
  hs[tid] = fmaxf(a, 0.0f);                       // ReLU
  __syncthreads();
  if (tid < NEXP) {
    float l = br2[tid];
    for (int j = 0; j < 256; ++j) l = fmaf(hs[j], Wr2[j * NEXP + tid], l);
    lg[tid] = l;
  }
  __syncthreads();
  if (tid == 0) {
    float mx = lg[0];
    for (int e = 1; e < NEXP; ++e) mx = fmaxf(mx, lg[e]);
    float g[NEXP], s = 0.0f;
    for (int e = 0; e < NEXP; ++e) { g[e] = __expf(lg[e] - mx); s += g[e]; }
    for (int e = 0; e < NEXP; ++e) g[e] /= s;     // softmax gates
    int e0 = 0;
    for (int e = 1; e < NEXP; ++e) if (g[e] > g[e0]) e0 = e;
    int e1 = (e0 == 0) ? 1 : 0;
    for (int e = 0; e < NEXP; ++e) if (e != e0 && g[e] > g[e1]) e1 = e;
    float a0 = __expf(g[e0] * 0.5f), a1 = __expf(g[e1] * 0.5f);  // softmax(tg/K)
    float inv = 1.0f / (a0 + a1);
    topk_e[t * 2 + 0] = e0;  topk_w[t * 2 + 0] = a0 * inv;
    topk_e[t * 2 + 1] = e1;  topk_w[t * 2 + 1] = a1 * inv;
    atomicAdd(&counts[e0], 1);
    atomicAdd(&counts[e1], 1);
  }
}

__global__ void k_scan(const int* __restrict__ counts, int* __restrict__ offsets,
                       int* __restrict__ counts2) {
  if (threadIdx.x == 0) {
    int s = 0;
    for (int e = 0; e < NEXP; ++e) { offsets[e] = s; s += counts[e]; }
  }
  if (threadIdx.x < NEXP) counts2[threadIdx.x] = 0;
}

__global__ void k_assign(const int* __restrict__ topk_e, int* __restrict__ counts2,
                         const int* __restrict__ offsets, int* __restrict__ slot_of,
                         int* __restrict__ tok_of_slot) {
  int t = blockIdx.x * 256 + threadIdx.x;
  #pragma unroll
  for (int k = 0; k < TOPK; ++k) {
    int e = topk_e[t * 2 + k];
    int slot = offsets[e] + atomicAdd(&counts2[e], 1);
    slot_of[t * 2 + k] = slot;
    tok_of_slot[slot] = t;
  }
}

__global__ __launch_bounds__(256)
void k_up_shared(const __bf16* __restrict__ xb, const float* __restrict__ Ws1,
                 const float* __restrict__ bs1, __bf16* __restrict__ Hs) {
  gemm128<false, true>(xb, nullptr, blockIdx.x * 128, 128,
                       Ws1, bs1, Hs, CDIM, IDIM, blockIdx.y * 128);
}

__global__ __launch_bounds__(256)
void k_up_routed(const __bf16* __restrict__ xb, const float* __restrict__ We1,
                 const float* __restrict__ be1, __bf16* __restrict__ Hr,
                 const int* __restrict__ tok_of_slot,
                 const int* __restrict__ offsets, const int* __restrict__ counts) {
  int e = blockIdx.z, rt = blockIdx.x;
  int cnt = counts[e];
  if (rt * 128 >= cnt) return;
  int nval = cnt - rt * 128; if (nval > 128) nval = 128;
  gemm128<true, true>(xb, tok_of_slot, offsets[e] + rt * 128, nval,
                      We1 + (size_t)e * CDIM * IDIM, be1 + e * IDIM,
                      Hr, CDIM, IDIM, blockIdx.y * 128);
}

__global__ __launch_bounds__(256)
void k_down_shared(const __bf16* __restrict__ Hs, const float* __restrict__ Ws2,
                   const float* __restrict__ bs2, float* __restrict__ out) {
  gemm128<false, false>(Hs, nullptr, blockIdx.x * 128, 128,
                        Ws2, bs2, out, IDIM, CDIM, blockIdx.y * 128);
}

__global__ __launch_bounds__(256)
void k_down_routed(const __bf16* __restrict__ Hr, const float* __restrict__ We2,
                   const float* __restrict__ be2, float* __restrict__ Rout,
                   const int* __restrict__ offsets, const int* __restrict__ counts) {
  int e = blockIdx.z, rt = blockIdx.x;
  int cnt = counts[e];
  if (rt * 128 >= cnt) return;
  int nval = cnt - rt * 128; if (nval > 128) nval = 128;
  gemm128<false, false>(Hr, nullptr, offsets[e] + rt * 128, nval,
                        We2 + (size_t)e * IDIM * CDIM, be2 + e * CDIM,
                        Rout, IDIM, CDIM, blockIdx.y * 128);
}

__global__ void k_combine(float* __restrict__ out, const float* __restrict__ Rout,
                          const int* __restrict__ slot_of,
                          const float* __restrict__ topk_w) {
  int t = blockIdx.y;
  int c = blockIdx.x * 256 + threadIdx.x;
  int   s0 = slot_of[t * 2],     s1 = slot_of[t * 2 + 1];
  float w0 = topk_w[t * 2],      w1 = topk_w[t * 2 + 1];
  out[(size_t)t * CDIM + c] += w0 * Rout[(size_t)s0 * CDIM + c]
                             + w1 * Rout[(size_t)s1 * CDIM + c];
}

// ---------------------------------------------------------------------------
// Workspace layout (~34.0 MiB total)
// ---------------------------------------------------------------------------
#define OFF_COUNTS   ((size_t)0)
#define OFF_COUNTS2  ((size_t)256)
#define OFF_OFFSETS  ((size_t)512)
#define OFF_TOPKE    ((size_t)768)
#define OFF_TOPKW    (OFF_TOPKE  + (size_t)NTOK * TOPK * 4)
#define OFF_SLOTOF   (OFF_TOPKW  + (size_t)NTOK * TOPK * 4)
#define OFF_TOKSLOT  (OFF_SLOTOF + (size_t)NTOK * TOPK * 4)
#define OFF_XB       (OFF_TOKSLOT + (size_t)NTOK * TOPK * 4)          // 2 MB
#define OFF_HS       (OFF_XB + (size_t)NTOK * CDIM * 2)               // 8 MB
#define OFF_HR       (OFF_HS + (size_t)NTOK * IDIM * 2)               // 16 MB
#define OFF_ROUT     (OFF_HR + (size_t)NTOK * TOPK * IDIM * 2)        // 8 MB

extern "C" void kernel_launch(void* const* d_in, const int* in_sizes, int n_in,
                              void* d_out, int out_size, void* d_ws, size_t ws_size,
                              hipStream_t stream) {
  const float* x   = (const float*)d_in[0];
  const float* Wr1 = (const float*)d_in[1];
  const float* br1 = (const float*)d_in[2];
  const float* Wr2 = (const float*)d_in[3];
  const float* br2 = (const float*)d_in[4];
  const float* Ws1 = (const float*)d_in[5];
  const float* bs1 = (const float*)d_in[6];
  const float* Ws2 = (const float*)d_in[7];
  const float* bs2 = (const float*)d_in[8];
  const float* We1 = (const float*)d_in[9];
  const float* be1 = (const float*)d_in[10];
  const float* We2 = (const float*)d_in[11];
  const float* be2 = (const float*)d_in[12];

  char* ws = (char*)d_ws;
  int*    counts  = (int*)   (ws + OFF_COUNTS);
  int*    counts2 = (int*)   (ws + OFF_COUNTS2);
  int*    offsets = (int*)   (ws + OFF_OFFSETS);
  int*    topk_e  = (int*)   (ws + OFF_TOPKE);
  float*  topk_w  = (float*) (ws + OFF_TOPKW);
  int*    slot_of = (int*)   (ws + OFF_SLOTOF);
  int*    tokslot = (int*)   (ws + OFF_TOKSLOT);
  __bf16* xb      = (__bf16*)(ws + OFF_XB);
  __bf16* Hs      = (__bf16*)(ws + OFF_HS);
  __bf16* Hr      = (__bf16*)(ws + OFF_HR);
  float*  Rout    = (float*) (ws + OFF_ROUT);
  float*  out     = (float*)d_out;

  k_init  <<<(NTOK * CDIM) / 256, 256, 0, stream>>>(x, xb, counts);
  k_router<<<NTOK, 256, 0, stream>>>(x, Wr1, br1, Wr2, br2, counts, topk_e, topk_w);
  k_scan  <<<1, 32, 0, stream>>>(counts, offsets, counts2);
  k_assign<<<NTOK / 256, 256, 0, stream>>>(topk_e, counts2, offsets, slot_of, tokslot);

  k_up_shared <<<dim3(NTOK / 128, IDIM / 128),        256, 0, stream>>>(xb, Ws1, bs1, Hs);
  k_up_routed <<<dim3(NTOK / 128, IDIM / 128, NEXP),  256, 0, stream>>>(xb, We1, be1, Hr,
                                                                        tokslot, offsets, counts);
  k_down_shared<<<dim3(NTOK / 128, CDIM / 128),       256, 0, stream>>>(Hs, Ws2, bs2, out);
  k_down_routed<<<dim3(NTOK / 128, CDIM / 128, NEXP), 256, 0, stream>>>(Hr, We2, be2, Rout,
                                                                        offsets, counts);
  k_combine<<<dim3(CDIM / 256, NTOK), 256, 0, stream>>>(out, Rout, slot_of, topk_w);
}
